// LSTMStack_45088566674139
// MI455X (gfx1250) — matrix operand order repaired and checked by reading, not compiled, not run on hardware
//
#include <hip/hip_runtime.h>
#include <stddef.h>
#include <stdint.h>

// ---------------------------------------------------------------------------
// 2-layer LSTM stack on gfx1250 (wave32, WMMA, async global->LDS staging).
// B=64, T=256, D=H=1024, gates=4H=4096.
// K is staged in 64-wide chunks (2 WMMA K-steps) to halve barrier/wait count.
// ---------------------------------------------------------------------------

typedef __bf16 bf16_t;
typedef __attribute__((ext_vector_type(16))) __bf16 v16bf;
typedef __attribute__((ext_vector_type(8)))  __bf16 v8bf;
typedef __attribute__((ext_vector_type(8)))  float  v8f;

#define LSTM_B   64
#define LSTM_T   256
#define LSTM_D   1024
#define LSTM_H   1024
#define LSTM_4H  4096
#define KCHUNK   64
#define NCHUNK   (LSTM_D / KCHUNK)   // 16

__device__ __forceinline__ float sigmoid_f(float x) {
    return 1.0f / (1.0f + __expf(-x));
}

// 16-byte async copy global -> LDS (GLOBAL_LOAD_ASYNC_TO_LDS_B128, ASYNCcnt).
__device__ __forceinline__ void async_copy_b128(uint32_t lds_off,
                                                const bf16_t* gsrc) {
    asm volatile("global_load_async_to_lds_b128 %0, %1, off"
                 :: "v"(lds_off), "v"(gsrc)
                 : "memory");
}

template <int N>
__device__ __forceinline__ void wait_asynccnt() {
    if constexpr (N == 0)      asm volatile("s_wait_asynccnt 0x0" ::: "memory");
    else if constexpr (N == 4) asm volatile("s_wait_asynccnt 0x4" ::: "memory");
    else                       asm volatile("s_wait_asynccnt 0x8" ::: "memory");
}

// A-fragment (16x32 bf16): lane reads contiguous K=[ka..ka+15].
__device__ __forceinline__ v16bf load_a_frag(const bf16_t* __restrict__ base,
                                             int row_stride, int m, int ka) {
    const bf16_t* p = base + (size_t)m * row_stride + ka;
    v8bf lo = *(const v8bf*)(p);
    v8bf hi = *(const v8bf*)(p + 8);
    return __builtin_shufflevector(lo, hi, 0,1,2,3,4,5,6,7,8,9,10,11,12,13,14,15);
}

// ---------------------------------------------------------------------------
// Fused LSTM cell. 4 waves per block = the 4 batch tiles; all waves share the
// same weight columns (n0..n0+15 in each of the 4 gate regions), so B-frags
// are staged once per block into LDS with async copies (double buffered).
//   DUAL=true : gates = A1@Wx + A2@Wh + b        (both weights staged)
//   DUAL=false: gates = xg_t + A2@Wh + b         (x@Wx precomputed)
// Grid: 64 blocks x 128 threads.
// ---------------------------------------------------------------------------
template <bool DUAL>
__global__ __launch_bounds__(128)
void lstm_cell_wmma(const bf16_t* __restrict__ A1, int lda1,   // DUAL only
                    const bf16_t* __restrict__ A2, int lda2,   // h_prev
                    const bf16_t* __restrict__ WxT,            // [4H,1024] (DUAL)
                    const bf16_t* __restrict__ WhT,            // [4H,1024]
                    const bf16_t* __restrict__ xg_t,           // [B,4H] (!DUAL)
                    const float*  __restrict__ bias,           // [4H]
                    float*        __restrict__ c_state,        // [B,H] in-place
                    bf16_t*       __restrict__ h_out,          // [B,H]
                    float*        __restrict__ h_out_f32)      // optional
{
    constexpr int ROWS = DUAL ? 128 : 64;   // staged LDS rows per buffer
    constexpr int NSEG = DUAL ? 8 : 4;      // async b128 issues per thread
    __shared__ bf16_t smem[2][ROWS][KCHUNK];   // 128B rows: [mat|gate|col][64 k]

    const int tid  = threadIdx.x;           // 0..127
    const int wave = tid >> 5;              // batch tile id
    const int lane = tid & 31;
    const int half = lane >> 4;
    const int ln   = lane & 15;
    const int m0   = wave << 4;
    const int n0   = (int)blockIdx.x << 4;
    const int m    = m0 + ln;
    const int ncol = n0 + ln;

    // Stage one 64-K chunk of B fragments into smem[buf].
    auto stage = [&](int buf, int kc) {
#pragma unroll
        for (int s = 0; s < NSEG; ++s) {
            const int q   = s * 128 + tid;      // [0, ROWS*8)
            const int row = q >> 3;
            const int seg = q & 7;
            const bf16_t* mat = (DUAL && row < 64) ? WxT : WhT;
            const int g = (row >> 4) & 3;
            const int c = row & 15;
            const bf16_t* src = mat +
                (size_t)(g * LSTM_H + n0 + c) * LSTM_D + kc + seg * 8;
            uint32_t dst = (uint32_t)(uintptr_t)&smem[buf][row][seg * 8];
            async_copy_b128(dst, src);
        }
    };

    // Accumulator init: bias (+ precomputed x@Wx tile in !DUAL mode).
    v8f acc[4];
#pragma unroll
    for (int g = 0; g < 4; ++g) {
        const float bv = bias[g * LSTM_H + ncol];
#pragma unroll
        for (int r = 0; r < 8; ++r) {
            float v = bv;
            if (!DUAL)
                v += (float)xg_t[(size_t)(m0 + 8 * half + r) * LSTM_4H +
                                 g * LSTM_H + ncol];
            acc[g][r] = v;
        }
    }

    stage(0, 0);
    for (int idx = 0; idx < NCHUNK; ++idx) {
        const int kc  = idx * KCHUNK;
        const int buf = idx & 1;
        if (idx + 1 < NCHUNK) {
            stage(buf ^ 1, kc + KCHUNK);
            wait_asynccnt<NSEG>();      // prior chunk done, prefetch in flight
        } else {
            wait_asynccnt<0>();
        }
        __syncthreads();

#pragma unroll
        for (int j = 0; j < 2; ++j) {               // two 32-K WMMA steps
            const int ks = kc + j * 32;
            v16bf ah = load_a_frag(A2, lda2, m, ks + 16 * half);
            v16bf ax;
            if (DUAL) ax = load_a_frag(A1, lda1, m, ks + 16 * half);

#pragma unroll
            for (int g = 0; g < 4; ++g) {
                if (DUAL) {
                    v16bf bx = *(const v16bf*)
                        &smem[buf][g * 16 + ln][j * 32 + half * 16];
                    acc[g] = __builtin_amdgcn_wmma_f32_16x16x32_bf16(
                                 false, ax, false, bx, (short)0, acc[g],
                                 false, false);
                }
                v16bf bh = *(const v16bf*)
                    &smem[buf][(DUAL ? 64 : 0) + g * 16 + ln][j * 32 + half * 16];
                acc[g] = __builtin_amdgcn_wmma_f32_16x16x32_bf16(
                             false, ah, false, bh, (short)0, acc[g],
                             false, false);
            }
        }
        __syncthreads();   // all waves done reading buf before it is restaged
    }

    // Fused LSTM element-wise update.
#pragma unroll
    for (int r = 0; r < 8; ++r) {
        const int mm = m0 + 8 * half + r;
        const size_t idx = (size_t)mm * LSTM_H + ncol;
        float iv = sigmoid_f(acc[0][r]);
        float fv = sigmoid_f(acc[1][r]);
        float gv = tanhf(acc[2][r]);
        float ov = sigmoid_f(acc[3][r]);
        float cn = fv * c_state[idx] + iv * gv;
        c_state[idx] = cn;
        float hn = ov * tanhf(cn);
        h_out[idx] = (bf16_t)hn;
        if (h_out_f32) h_out_f32[idx] = hn;
    }
}

// ---------------------------------------------------------------------------
// One-time parallel GEMM: xg[t,b,:] = x[b,t,:] @ Wx0   (bf16 out, no bias).
// Same LDS-staging structure; grid (64 col-blocks, 256 t) x 128 threads.
// ---------------------------------------------------------------------------
__global__ __launch_bounds__(128)
void gemm_x_wx0(const bf16_t* __restrict__ x_bf,    // [B,T,D] bf16
                const bf16_t* __restrict__ WxT,     // [4H,1024] bf16
                bf16_t*       __restrict__ xg)      // [T,B,4H] bf16
{
    __shared__ bf16_t smem[2][64][KCHUNK];

    const int tid  = threadIdx.x;
    const int wave = tid >> 5;
    const int lane = tid & 31;
    const int half = lane >> 4;
    const int ln   = lane & 15;
    const int m0   = wave << 4;
    const int n0   = (int)blockIdx.x << 4;
    const int t    = (int)blockIdx.y;
    const int m    = m0 + ln;
    const int ncol = n0 + ln;

    const bf16_t* A = x_bf + (size_t)t * LSTM_D;     // row stride T*D

    auto stage = [&](int buf, int kc) {
#pragma unroll
        for (int s = 0; s < 4; ++s) {
            const int q   = s * 128 + tid;
            const int row = q >> 3;
            const int seg = q & 7;
            const int g = (row >> 4) & 3;
            const int c = row & 15;
            const bf16_t* src = WxT +
                (size_t)(g * LSTM_H + n0 + c) * LSTM_D + kc + seg * 8;
            uint32_t dst = (uint32_t)(uintptr_t)&smem[buf][row][seg * 8];
            async_copy_b128(dst, src);
        }
    };

    v8f acc[4];
#pragma unroll
    for (int g = 0; g < 4; ++g)
#pragma unroll
        for (int r = 0; r < 8; ++r) acc[g][r] = 0.0f;

    stage(0, 0);
    for (int idx = 0; idx < NCHUNK; ++idx) {
        const int kc  = idx * KCHUNK;
        const int buf = idx & 1;
        if (idx + 1 < NCHUNK) {
            stage(buf ^ 1, kc + KCHUNK);
            wait_asynccnt<4>();
        } else {
            wait_asynccnt<0>();
        }
        __syncthreads();

#pragma unroll
        for (int j = 0; j < 2; ++j) {
            const int ks = kc + j * 32;
            v16bf ax = load_a_frag(A, LSTM_T * LSTM_D, m, ks + 16 * half);
#pragma unroll
            for (int g = 0; g < 4; ++g) {
                v16bf bx = *(const v16bf*)
                    &smem[buf][g * 16 + ln][j * 32 + half * 16];
                acc[g] = __builtin_amdgcn_wmma_f32_16x16x32_bf16(
                             false, ax, false, bx, (short)0, acc[g],
                             false, false);
            }
        }
        __syncthreads();
    }

#pragma unroll
    for (int g = 0; g < 4; ++g)
#pragma unroll
        for (int r = 0; r < 8; ++r) {
            const int mm = m0 + 8 * half + r;
            xg[((size_t)t * LSTM_B + mm) * LSTM_4H + g * LSTM_H + ncol] =
                (bf16_t)acc[g][r];
        }
}

// ---------------------------------------------------------------------------
// Prep kernels.
// ---------------------------------------------------------------------------
__global__ void transpose_conv_bf16(const float* __restrict__ in,
                                    bf16_t* __restrict__ out, int K, int N)
{
    int idx = blockIdx.x * blockDim.x + threadIdx.x;
    if (idx >= K * N) return;
    int k = idx / N;
    int n = idx - k * N;
    out[(size_t)n * K + k] = (bf16_t)in[idx];
}

__global__ void conv_bf16(const float* __restrict__ in,
                          bf16_t* __restrict__ out, int n)
{
    int i = blockIdx.x * blockDim.x + threadIdx.x;
    if (i < n) out[i] = (bf16_t)in[i];
}

__global__ void zero_bf16(bf16_t* __restrict__ p, int n)
{
    int i = blockIdx.x * blockDim.x + threadIdx.x;
    if (i < n) p[i] = (bf16_t)0.0f;
}

__global__ void zero_f32(float* __restrict__ p, int n)
{
    int i = blockIdx.x * blockDim.x + threadIdx.x;
    if (i < n) p[i] = 0.0f;
}

// ---------------------------------------------------------------------------
// Host-side orchestration.
// ---------------------------------------------------------------------------
extern "C" void kernel_launch(void* const* d_in, const int* in_sizes, int n_in,
                              void* d_out, int out_size, void* d_ws, size_t ws_size,
                              hipStream_t stream)
{
    (void)in_sizes; (void)n_in; (void)out_size;

    const float* x   = (const float*)d_in[0];
    const float* Wx0 = (const float*)d_in[1];
    const float* Wh0 = (const float*)d_in[2];
    const float* b0  = (const float*)d_in[3];
    const float* Wx1 = (const float*)d_in[4];
    const float* Wh1 = (const float*)d_in[5];
    const float* b1  = (const float*)d_in[6];
    float* out = (float*)d_out;

    const size_t W_ELEMS = (size_t)LSTM_D * LSTM_4H;           // 4M
    const size_t W_BYTES = W_ELEMS * sizeof(bf16_t);           // 8 MB
    const size_t X_ELEMS = (size_t)LSTM_B * LSTM_T * LSTM_D;   // 16.7M
    const size_t S_ELEMS = (size_t)LSTM_B * LSTM_H;            // 64K
    const size_t XG_ELEMS = (size_t)LSTM_T * LSTM_B * LSTM_4H; // 67M (128 MB bf16)

    char* ws = (char*)d_ws;
    bf16_t* WxT0 = (bf16_t*)(ws);
    bf16_t* WhT0 = (bf16_t*)(ws + 1 * W_BYTES);
    bf16_t* WxT1 = (bf16_t*)(ws + 2 * W_BYTES);
    bf16_t* WhT1 = (bf16_t*)(ws + 3 * W_BYTES);
    bf16_t* x_bf = (bf16_t*)(ws + 4 * W_BYTES);
    char* p = ws + 4 * W_BYTES + X_ELEMS * sizeof(bf16_t);
    bf16_t* h0_buf[2] = { (bf16_t*)p, (bf16_t*)(p + S_ELEMS * sizeof(bf16_t)) };
    p += 2 * S_ELEMS * sizeof(bf16_t);
    bf16_t* h1_buf[2] = { (bf16_t*)p, (bf16_t*)(p + S_ELEMS * sizeof(bf16_t)) };
    p += 2 * S_ELEMS * sizeof(bf16_t);
    float* c0 = (float*)p;             p += S_ELEMS * sizeof(float);
    float* c1 = (float*)p;             p += S_ELEMS * sizeof(float);
    const size_t base_bytes = (size_t)(p - ws);

    // Optional precompute buffer for xg = x @ Wx0 (gated on ws_size).
    const bool use_pre = ws_size >= base_bytes + XG_ELEMS * sizeof(bf16_t);
    bf16_t* xg = (bf16_t*)p;

    // --- prep ---------------------------------------------------------------
    {
        const int thr = 256;
        int wblk = (int)((W_ELEMS + thr - 1) / thr);
        transpose_conv_bf16<<<wblk, thr, 0, stream>>>(Wx0, WxT0, LSTM_D, LSTM_4H);
        transpose_conv_bf16<<<wblk, thr, 0, stream>>>(Wh0, WhT0, LSTM_H, LSTM_4H);
        transpose_conv_bf16<<<wblk, thr, 0, stream>>>(Wx1, WxT1, LSTM_H, LSTM_4H);
        transpose_conv_bf16<<<wblk, thr, 0, stream>>>(Wh1, WhT1, LSTM_H, LSTM_4H);

        int xblk = (int)((X_ELEMS + thr - 1) / thr);
        conv_bf16<<<xblk, thr, 0, stream>>>(x, x_bf, (int)X_ELEMS);

        int sblk = (int)((S_ELEMS + thr - 1) / thr);
        zero_bf16<<<sblk, thr, 0, stream>>>(h0_buf[0], (int)S_ELEMS);
        zero_bf16<<<sblk, thr, 0, stream>>>(h1_buf[0], (int)S_ELEMS);
        zero_f32 <<<sblk, thr, 0, stream>>>(c0, (int)S_ELEMS);
        zero_f32 <<<sblk, thr, 0, stream>>>(c1, (int)S_ELEMS);
    }

    // --- hoisted parallel GEMM: xg = x @ Wx0 (no sequential dependency) -----
    if (use_pre) {
        gemm_x_wx0<<<dim3(LSTM_4H / 64, LSTM_T), 128, 0, stream>>>(x_bf, WxT0, xg);
    }

    // --- recurrent chain ----------------------------------------------------
    const int blocks = LSTM_H / 16;    // 64 blocks x 4 waves
    for (int t = 0; t < LSTM_T; ++t) {
        const int pi = t & 1, qi = pi ^ 1;
        if (use_pre) {
            // Layer 0: gates = xg[t] + h0_prev @ Wh0 + b0
            lstm_cell_wmma<false><<<blocks, 128, 0, stream>>>(
                nullptr, 0,
                h0_buf[pi], LSTM_H,
                nullptr, WhT0,
                xg + (size_t)t * LSTM_B * LSTM_4H,
                b0, c0, h0_buf[qi], nullptr);
        } else {
            // Layer 0 fused: gates = x_t @ Wx0 + h0_prev @ Wh0 + b0
            lstm_cell_wmma<true><<<blocks, 128, 0, stream>>>(
                x_bf + (size_t)t * LSTM_D, LSTM_T * LSTM_D,
                h0_buf[pi], LSTM_H,
                WxT0, WhT0, nullptr,
                b0, c0, h0_buf[qi], nullptr);
        }
        // Layer 1: gates = h0_new @ Wx1 + h1_prev @ Wh1 + b1
        lstm_cell_wmma<true><<<blocks, 128, 0, stream>>>(
            h0_buf[qi], LSTM_H,
            h1_buf[pi], LSTM_H,
            WxT1, WhT1, nullptr,
            b1, c1, h1_buf[qi], (t == LSTM_T - 1) ? out : nullptr);
    }
}
